// Attention_47132971106602
// MI455X (gfx1250) — compile-verified, matrix operand order checked by reading, not run on hardware
//
#include <hip/hip_runtime.h>

// ---------------- problem constants ----------------
#define HID   2048
#define NH    16
#define DH    128
#define QKVC  6144      // 3*NH*DH
#define PROJC 8192      // QKVC + HID
#define SEQL  2048
#define NB    4

typedef __attribute__((ext_vector_type(16))) __bf16 v16bf;
typedef __attribute__((ext_vector_type(8)))  float  v8f;

// fp32 -> bf16 round-to-nearest-even
__device__ inline __bf16 f2bf(float f) {
  unsigned u = __builtin_bit_cast(unsigned, f);
  unsigned r = (u + 0x7FFFu + ((u >> 16) & 1u)) >> 16;
  return __builtin_bit_cast(__bf16, (unsigned short)r);
}
__device__ inline __bf16 cvt_bf(float x)  { return f2bf(x); }
__device__ inline __bf16 cvt_bf(__bf16 x) { return x; }

__device__ inline v8f wmma_bf16(v16bf a, v16bf b, v8f c) {
  // (neg_a, A, neg_b, B, c_mod, C, reuse_a, reuse_b)
  return __builtin_amdgcn_wmma_f32_16x16x32_bf16(false, a, false, b, (short)0, c, false, false);
}

// ---- CDNA5 async global->LDS copy (ASYNCcnt path), 16B per lane ----
__device__ inline unsigned lds_off(const void* p) {
  // LDS aperture: hardware uses addr[31:0] as the LDS byte offset
  return (unsigned)(size_t)p;
}
__device__ inline void async_ld_b128(unsigned ldsoff, const void* gaddr) {
  asm volatile("global_load_async_to_lds_b128 %0, %1, off"
               :: "v"(ldsoff), "v"(gaddr) : "memory");
}
__device__ inline void wait_async() {
  asm volatile("s_wait_asynccnt 0" ::: "memory");
}

// Load a 16x32 bf16 A-fragment (or B^T-fragment) from row-major LDS
// [rows][stride], rows r0..r0+15, K-columns k0..k0+31.
// ISA layout: lane l row = l&15 ; elem i<8 -> K = (l>=16?8:0)+i ;
//             elem i>=8 -> K = 16+(l>=16?8:0)+(i-8).
__device__ inline v16bf load_frag(const __bf16* sm, int stride, int r0, int k0) {
  const int lane = threadIdx.x & 31;
  const __bf16* p = sm + (r0 + (lane & 15)) * stride + k0 + ((lane >> 4) << 3);
  v16bf f;
#pragma unroll
  for (int i = 0; i < 8; ++i) f[i] = p[i];
#pragma unroll
  for (int i = 0; i < 8; ++i) f[i + 8] = p[i + 16];
  return f;
}

// ============ generic tiled GEMM: C(f32) = A @ B(f32), bf16 WMMA ============
// block tile 128x128, K-step 32, 256 threads = 8 waves (2x4 wave grid),
// wave tile 64x32 = 4x2 WMMA fragments.
#define LDST 40  // padded LDS row stride: 80B = 20 banks; 16B-aligned rows

template <typename AT>
__global__ __launch_bounds__(256) void gemm_wmma_kernel(
    const AT* __restrict__ A, const float* __restrict__ B, float* __restrict__ C,
    int M, int N, int K)
{
  __shared__ alignas(16) __bf16 As[128 * LDST];
  __shared__ alignas(16) __bf16 Bs[128 * LDST];  // stored transposed: Bs[n][k]

  const int n0  = blockIdx.x * 128;
  const int m0  = blockIdx.y * 128;
  const int t   = threadIdx.x;
  const int wid = t >> 5;
  const int wm  = wid >> 2;  // 0..1
  const int wn  = wid & 3;   // 0..3

  v8f acc[4][2];
#pragma unroll
  for (int i = 0; i < 4; ++i)
#pragma unroll
    for (int j = 0; j < 2; ++j)
#pragma unroll
      for (int r = 0; r < 8; ++r) acc[i][j][r] = 0.f;

  for (int k0 = 0; k0 < K; k0 += 32) {
    // A tile: 128 rows x 32 K
    if constexpr (__is_same(AT, __bf16)) {
      // straight bf16 copy -> async DMA to LDS (512 x 16B chunks, 2/thread)
#pragma unroll
      for (int p = 0; p < 2; ++p) {
        int e = p * 256 + t;
        int r = e >> 2, c = e & 3;
        async_ld_b128(lds_off(&As[r * LDST + c * 8]),
                      &A[(size_t)(m0 + r) * K + k0 + c * 8]);
      }
    } else {
      // fp32 -> bf16 convert through VGPRs
#pragma unroll
      for (int p = 0; p < 4; ++p) {
        int r = p * 32 + (t >> 3);
        int c = (t & 7) * 4;
        const AT* s = A + (size_t)(m0 + r) * K + (k0 + c);
        __bf16* d = As + r * LDST + c;
#pragma unroll
        for (int i = 0; i < 4; ++i) d[i] = cvt_bf(s[i]);
      }
    }
    // B tile: 32 K x 128 n (fp32), store transposed Bs[n][k]
#pragma unroll
    for (int p = 0; p < 4; ++p) {
      int k = p * 8 + (t >> 5);
      int n = (t & 31) * 4;
      const float* s = B + (size_t)(k0 + k) * N + (n0 + n);
#pragma unroll
      for (int i = 0; i < 4; ++i) Bs[(n + i) * LDST + k] = f2bf(s[i]);
    }
    if constexpr (__is_same(AT, __bf16)) wait_async();
    __syncthreads();

    v16bf bfr0 = load_frag(Bs, LDST, wn * 32,      0);
    v16bf bfr1 = load_frag(Bs, LDST, wn * 32 + 16, 0);
#pragma unroll
    for (int im = 0; im < 4; ++im) {
      v16bf af = load_frag(As, LDST, wm * 64 + im * 16, 0);
      acc[im][0] = wmma_bf16(af, bfr0, acc[im][0]);
      acc[im][1] = wmma_bf16(af, bfr1, acc[im][1]);
    }
    __syncthreads();
  }

  const int lane  = t & 31;
  const int halfo = (lane >> 4) << 3;
  const int col   = lane & 15;
#pragma unroll
  for (int im = 0; im < 4; ++im)
#pragma unroll
    for (int jn = 0; jn < 2; ++jn)
#pragma unroll
      for (int r = 0; r < 8; ++r) {
        int row = m0 + wm * 64 + im * 16 + halfo + r;
        int cc  = n0 + wn * 32 + jn * 16 + col;
        C[(size_t)row * N + cc] = acc[im][jn][r];
      }
}

// ============ kernel 2: rmsnorm + rope + pack q/k/v to bf16 head-major ======
__device__ inline void norm_rope_store(const float* __restrict__ src,
                                       __bf16* __restrict__ dst, int pos) {
  const int lane = threadIdx.x & 31;     // each lane owns 4 of DH=128 elems
  float x[4];
  float ssq = 0.f;
#pragma unroll
  for (int i = 0; i < 4; ++i) { x[i] = src[lane * 4 + i]; ssq += x[i] * x[i]; }
#pragma unroll
  for (int off = 16; off >= 1; off >>= 1) ssq += __shfl_xor(ssq, off, 32);
  const float rn = rsqrtf(ssq * (1.f / 128.f) + 1e-5f);
  const bool hi = lane >= 16;            // lanes>=16 own d in [64,128)
#pragma unroll
  for (int i = 0; i < 4; ++i) {
    float xn = x[i] * rn;
    float partner = __shfl_xor(xn, 16, 32);  // pairs d <-> d+64
    int d = lane * 4 + i;
    int j = hi ? d - 64 : d;
    float inv_freq = __expf(-(float)j * (9.210340371976184f / 64.f)); // 10000^(-j/64)
    float ang = (float)pos * inv_freq;
    float s, c;
    __sincosf(ang, &s, &c);
    float out = hi ? (partner * s + xn * c) : (xn * c - partner * s);
    dst[d] = f2bf(out);
  }
}

__global__ __launch_bounds__(256) void rope_pack_kernel(
    const float* __restrict__ proj, __bf16* __restrict__ qh,
    __bf16* __restrict__ kh, __bf16* __restrict__ vh)
{
  const int tok  = blockIdx.x;           // 0..B*L-1
  const int b    = tok >> 11;
  const int pos  = tok & (SEQL - 1);
  const int w    = threadIdx.x >> 5;
  const int lane = threadIdx.x & 31;
  const float* base = proj + (size_t)tok * PROJC;
  for (int h = w; h < NH; h += 8) {
    size_t hoff = (((size_t)b * NH + h) * SEQL + pos) * DH;
    norm_rope_store(base + h * DH,        qh + hoff, pos);
    norm_rope_store(base + HID + h * DH,  kh + hoff, pos);
    const float* vs = base + 2 * HID + h * DH;
    __bf16* vd = vh + hoff;
#pragma unroll
    for (int i = 0; i < 4; ++i) vd[lane * 4 + i] = f2bf(vs[lane * 4 + i]);
  }
}

// ============ kernel 3: flash attention + gate, WMMA QK^T and PV ============
#define KST 136  // stride for [*][DH] LDS tiles: 272B rows, 16B-aligned
#define VST 34   // stride for [*][32] LDS tiles (scalar b16 writes)

__global__ __launch_bounds__(128) void attn_kernel(
    const __bf16* __restrict__ qh, const __bf16* __restrict__ kh,
    const __bf16* __restrict__ vh, const float* __restrict__ proj,
    __bf16* __restrict__ attnout)
{
  __shared__ alignas(16) __bf16 Qsm[64 * KST];       // [qrow][dh]
  __shared__ alignas(16) __bf16 Ksm[32 * KST];       // [key][dh]   (= B^T for QK^T)
  __shared__ alignas(16) __bf16 Vsm[128 * VST];      // [dh][key]   (= B^T for PV)
  __shared__ alignas(16) __bf16 Psm[4 * 16 * VST];   // per-wave P (16x32)

  const int q0 = blockIdx.x * 64;
  const int h  = blockIdx.y;
  const int b  = blockIdx.z;
  const int t  = threadIdx.x;
  const int w  = t >> 5;
  const int lane  = t & 31;
  const int halfo = (lane >> 4) << 3;
  const int col   = lane & 15;

  const __bf16* Qg = qh + (((size_t)b * NH + h) * SEQL + q0) * DH;
  const __bf16* Kg = kh + (((size_t)b * NH + h) * SEQL) * DH;
  const __bf16* Vg = vh + (((size_t)b * NH + h) * SEQL) * DH;

  // Q tile 64x128 -> LDS via async DMA (512 x 16B chunks)
  for (int e = t; e < 64 * 8; e += 128) {
    int row = e >> 3, ch = e & 7;
    async_ld_b128(lds_off(&Qsm[row * KST + ch * 8]), &Qg[row * DH + ch * 8]);
  }
  wait_async();
  __syncthreads();

  v16bf qf[4];
#pragma unroll
  for (int c = 0; c < 4; ++c) qf[c] = load_frag(Qsm, KST, w * 16, c * 32);

  v8f accO[8];
  float mrow[8], lrow[8];
#pragma unroll
  for (int f = 0; f < 8; ++f)
#pragma unroll
    for (int r = 0; r < 8; ++r) accO[f][r] = 0.f;
#pragma unroll
  for (int r = 0; r < 8; ++r) { mrow[r] = -1e30f; lrow[r] = 0.f; }

  const int nkb = (q0 >> 5) + 2;                 // causal: keys up to q0+63
  const float scale = 0.08838834764831845f;      // 1/sqrt(128)

  for (int kb = 0; kb < nkb; ++kb) {
    __syncthreads();  // previous PV reads done before overwriting tiles
    const __bf16* Kb = Kg + (size_t)kb * 32 * DH;
    const __bf16* Vb = Vg + (size_t)kb * 32 * DH;
    // K tile: straight copy -> async DMA (256 x 16B chunks)
    for (int e = t; e < 32 * 8; e += 128) {
      int row = e >> 3, ch = e & 7;
      async_ld_b128(lds_off(&Ksm[row * KST + ch * 8]), &Kb[row * DH + ch * 8]);
    }
    // V tile: transpose into Vsm[dh][key] (VGPR path, scalar b16 stores)
    for (int e = t; e < 32 * 64; e += 128) {
      int key = e >> 6, dh2 = e & 63;
      unsigned u = *(const unsigned*)(&Vb[key * DH + dh2 * 2]);
      Vsm[(dh2 * 2)     * VST + key] = __builtin_bit_cast(__bf16, (unsigned short)(u & 0xFFFFu));
      Vsm[(dh2 * 2 + 1) * VST + key] = __builtin_bit_cast(__bf16, (unsigned short)(u >> 16));
    }
    wait_async();
    __syncthreads();

    // S = Q(16x128) @ K^T(128x32): K-dim in 4 chunks of 32
    v8f s0, s1;
#pragma unroll
    for (int r = 0; r < 8; ++r) { s0[r] = 0.f; s1[r] = 0.f; }
#pragma unroll
    for (int c = 0; c < 4; ++c) {
      v16bf kf0 = load_frag(Ksm, KST, 0,  c * 32);
      v16bf kf1 = load_frag(Ksm, KST, 16, c * 32);
      s0 = wmma_bf16(qf[c], kf0, s0);
      s1 = wmma_bf16(qf[c], kf1, s1);
    }

    // online softmax in the C-layout registers
    __bf16* pw = Psm + w * (16 * VST);
    const int qbase = q0 + w * 16 + halfo;
    const int kbase = kb * 32;
#pragma unroll
    for (int r = 0; r < 8; ++r) {
      int qg2 = qbase + r;
      float a  = s0[r] * scale; if (kbase + col      > qg2) a  = -1e30f;
      float bb = s1[r] * scale; if (kbase + 16 + col > qg2) bb = -1e30f;
      float m = fmaxf(a, bb);
#pragma unroll
      for (int off = 8; off >= 1; off >>= 1) m = fmaxf(m, __shfl_xor(m, off, 16));
      float mnew = fmaxf(mrow[r], m);
      float corr = __expf(mrow[r] - mnew);
      float p0 = __expf(a  - mnew);
      float p1 = __expf(bb - mnew);
      float rs = p0 + p1;
#pragma unroll
      for (int off = 8; off >= 1; off >>= 1) rs += __shfl_xor(rs, off, 16);
      lrow[r] = lrow[r] * corr + rs;
      mrow[r] = mnew;
#pragma unroll
      for (int f = 0; f < 8; ++f) accO[f][r] *= corr;
      pw[(halfo + r) * VST + col]      = f2bf(p0);
      pw[(halfo + r) * VST + 16 + col] = f2bf(p1);
    }
    __syncthreads();  // make cross-lane P stores visible before A-frag reload

    // O += P(16x32) @ V(32x128)
    v16bf pf = load_frag(pw, VST, 0, 0);
#pragma unroll
    for (int f = 0; f < 8; ++f) {
      v16bf vf = load_frag(Vsm, VST, f * 16, 0);
      accO[f] = wmma_bf16(pf, vf, accO[f]);
    }
  }

  // epilogue: normalize, apply sigmoid gate (from proj tail cols), store bf16
#pragma unroll
  for (int f = 0; f < 8; ++f) {
#pragma unroll
    for (int r = 0; r < 8; ++r) {
      int q  = q0 + w * 16 + halfo + r;
      int dh = f * 16 + col;
      float o = accO[f][r] / lrow[r];
      float g = proj[((size_t)b * SEQL + q) * PROJC + QKVC + h * DH + dh];
      g = 1.f / (1.f + __expf(-g));
      attnout[((size_t)b * SEQL + q) * HID + h * DH + dh] = f2bf(o * g);
    }
  }
}

// ============================ host launcher ================================
extern "C" void kernel_launch(void* const* d_in, const int* in_sizes, int n_in,
                              void* d_out, int out_size, void* d_ws, size_t ws_size,
                              hipStream_t stream) {
  (void)in_sizes; (void)n_in; (void)out_size; (void)ws_size;
  const float* X     = (const float*)d_in[0];   // (4, 2048, 2048)
  const float* Wqkvg = (const float*)d_in[1];   // (2048, 8192)
  const float* Wout  = (const float*)d_in[2];   // (2048, 2048)
  float* out = (float*)d_out;                   // (4, 2048, 2048)

  const size_t M = (size_t)NB * SEQL;           // 8192 tokens
  char* ws = (char*)d_ws;
  float*  proj    = (float*)ws;                                  // 8192*8192 f32 (256MB)
  __bf16* qh      = (__bf16*)(ws + M * PROJC * sizeof(float));   // [B][H][L][DH]
  __bf16* kh      = qh + M * HID;
  __bf16* vh      = kh + M * HID;
  __bf16* attnout = vh + M * HID;                                // [B*L][HID] bf16

  // 1) proj = X @ W_qkvg   (8192 x 8192, K=2048)
  gemm_wmma_kernel<float><<<dim3(PROJC / 128, M / 128), 256, 0, stream>>>(
      X, Wqkvg, proj, (int)M, PROJC, HID);

  // 2) rmsnorm + rope + pack q/k/v (bf16 head-major)
  rope_pack_kernel<<<(int)M, 256, 0, stream>>>(proj, qh, kh, vh);

  // 3) causal flash attention + sigmoid gate
  attn_kernel<<<dim3(SEQL / 64, NH, NB), 128, 0, stream>>>(qh, kh, vh, proj, attnout);

  // 4) out = attnout @ W_out   (8192 x 2048, K=2048)
  gemm_wmma_kernel<__bf16><<<dim3(HID / 128, M / 128), 256, 0, stream>>>(
      attnout, Wout, out, (int)M, HID, HID);
}
